// Encoder1_46763603919350
// MI455X (gfx1250) — matrix operand order repaired and checked by reading, not compiled, not run on hardware
//
#include <hip/hip_runtime.h>
#include <hip/hip_bf16.h>

// Problem constants (from reference)
#define NNODES 50000
#define NEDGES 600000
#define DIN    128
#define DH     128

typedef __attribute__((ext_vector_type(16))) _Float16 v16h;
typedef __attribute__((ext_vector_type(8)))  _Float16 v8h;
typedef __attribute__((ext_vector_type(8)))  float    v8f;

// ---------------------------------------------------------------------------
// 1) Convert x [N,DIN] and W [DH,DIN] from f32 to f16 (row-major, contiguous)
// ---------------------------------------------------------------------------
__global__ void convert_f16_kernel(const float* __restrict__ x,
                                   const float* __restrict__ W,
                                   _Float16* __restrict__ xh,
                                   _Float16* __restrict__ wh) {
    int tid = blockIdx.x * blockDim.x + threadIdx.x;
    const int nx = NNODES * DIN;
    if (tid < nx) {
        xh[tid] = (_Float16)x[tid];
    } else {
        int t = tid - nx;
        if (t < DH * DIN) wh[t] = (_Float16)W[t];
    }
}

// ---------------------------------------------------------------------------
// 2) h = xh @ whT  via v_wmma_f32_16x16x32_f16.
//    One wave -> one 16x16 output tile. Block = 8 waves = one M-tile row
//    (8 N-tiles of 16 cover DH=128). Grid = N/16 = 3125 blocks (exact).
//    Fragment layouts per CDNA5 ISA 7.12.2 (wave32):
//      A (16x32 f16):  lane<16: row=lane,    K = {k0..k0+7, k0+16..k0+23}
//                      lane>=16: row=lane-16, K = {k0+8..k0+15, k0+24..k0+31}
//      B (32x16 f16):  lane<16: col=lane,    K = {k0..k0+7, k0+16..k0+23}
//                      lane>=16: col=lane-16, K = {k0+8..k0+15, k0+24..k0+31}
//      C/D (16x16 f32): lane<16: N=lane, VGPR v -> M=v; lane>=16: M=v+8
// ---------------------------------------------------------------------------
__global__ void gemm_wmma_kernel(const _Float16* __restrict__ xh,
                                 const _Float16* __restrict__ wh,
                                 float* __restrict__ h) {
    const int wave  = threadIdx.x >> 5;   // 0..7  -> N tile
    const int lane  = threadIdx.x & 31;
    const int lhalf = lane & 15;
    const bool hi   = lane >= 16;

    const int m0 = blockIdx.x * 16;
    const int n0 = wave * 16;

    const _Float16* arow = xh + (size_t)(m0 + lhalf) * DIN; // x row for this lane
    const _Float16* brow = wh + (size_t)(n0 + lhalf) * DIN; // W row == B column

    v8f c = {};
    union { v16h v; v8h p[2]; } a, b;

#pragma unroll
    for (int k0 = 0; k0 < DIN; k0 += 32) {
        const int ka = k0 + (hi ? 8 : 0);
        a.p[0] = *(const v8h*)(arow + ka);        // 16B contiguous
        a.p[1] = *(const v8h*)(arow + ka + 16);   // 16B contiguous
        const int kb = k0 + (hi ? 8 : 0);
        b.p[0] = *(const v8h*)(brow + kb);
        b.p[1] = *(const v8h*)(brow + kb + 16);
        c = __builtin_amdgcn_wmma_f32_16x16x32_f16(
                /*neg_a=*/false, a.v, /*neg_b=*/false, b.v,
                /*c_mod=*/(short)0, c, /*reuse_a=*/false, /*reuse_b=*/false);
    }

    float* hptr = h + (size_t)(m0 + (hi ? 8 : 0)) * DH + (n0 + lhalf);
#pragma unroll
    for (int v = 0; v < 8; ++v) hptr[(size_t)v * DH] = c[v];
}

// ---------------------------------------------------------------------------
// 3) Degree: init to 1.0 (self-loop weight), scatter-add edge weights,
//    then in-place d^-1/2.
// ---------------------------------------------------------------------------
__global__ void deg_init_kernel(float* __restrict__ deg) {
    int i = blockIdx.x * blockDim.x + threadIdx.x;
    if (i < NNODES) deg[i] = 1.0f;
}

__global__ void deg_scatter_kernel(const long long* __restrict__ ei,
                                   const float* __restrict__ w,
                                   float* __restrict__ deg) {
    int e = blockIdx.x * blockDim.x + threadIdx.x;
    if (e < NEDGES) {
        int d = (int)ei[NEDGES + e];   // dst row of edge_index [2,E]
        atomicAdd(&deg[d], w[e]);
    }
}

__global__ void dinv_kernel(float* __restrict__ deg) {
    int i = blockIdx.x * blockDim.x + threadIdx.x;
    if (i < NNODES) {
        float d = deg[i];
        deg[i] = (d > 0.0f) ? rsqrtf(d) : 0.0f;   // deg >= 1 always, kept for safety
    }
}

// ---------------------------------------------------------------------------
// 4) out init: bias + self-loop term (norm for self loop = dinv[i]^2, w=1).
//    Non-atomic full overwrite (d_out is poisoned by the harness).
// ---------------------------------------------------------------------------
__global__ void out_init_kernel(const float* __restrict__ h,
                                const float* __restrict__ dinv,
                                const float* __restrict__ b,
                                float* __restrict__ out) {
    int tid = blockIdx.x * blockDim.x + threadIdx.x;
    if (tid < NNODES * DH) {
        int i = tid >> 7;          // node
        int j = tid & (DH - 1);    // channel
        float dv = dinv[i];
        out[tid] = b[j] + dv * dv * h[tid];
    }
}

// ---------------------------------------------------------------------------
// 5) Edge aggregation: one wave per edge; lane covers 4 channels (float4).
//    Gather h[src] (128B/lane-quad coalesced, L2-resident), scatter-add into
//    out[dst] with global f32 atomics.
// ---------------------------------------------------------------------------
__global__ void edge_scatter_kernel(const long long* __restrict__ ei,
                                    const float* __restrict__ w,
                                    const float* __restrict__ dinv,
                                    const float* __restrict__ h,
                                    float* __restrict__ out) {
    int gid  = blockIdx.x * blockDim.x + threadIdx.x;
    int e    = gid >> 5;
    int lane = gid & 31;
    if (e >= NEDGES) return;

    int s = (int)ei[e];
    int d = (int)ei[NEDGES + e];
    float nrm = dinv[s] * w[e] * dinv[d];

    float4 hv = ((const float4*)(h + (size_t)s * DH))[lane];
    float* op = out + (size_t)d * DH + lane * 4;
    atomicAdd(op + 0, nrm * hv.x);
    atomicAdd(op + 1, nrm * hv.y);
    atomicAdd(op + 2, nrm * hv.z);
    atomicAdd(op + 3, nrm * hv.w);
}

// ---------------------------------------------------------------------------
// 6) PReLU epilogue (per-channel slope)
// ---------------------------------------------------------------------------
__global__ void prelu_kernel(const float* __restrict__ alpha,
                             float* __restrict__ out) {
    int tid = blockIdx.x * blockDim.x + threadIdx.x;
    if (tid < NNODES * DH) {
        int j = tid & (DH - 1);
        float v = out[tid];
        out[tid] = (v > 0.0f) ? v : alpha[j] * v;
    }
}

// ---------------------------------------------------------------------------
// launcher
// ---------------------------------------------------------------------------
extern "C" void kernel_launch(void* const* d_in, const int* in_sizes, int n_in,
                              void* d_out, int out_size, void* d_ws, size_t ws_size,
                              hipStream_t stream) {
    const float*     x     = (const float*)d_in[0];
    const long long* ei    = (const long long*)d_in[1];  // edge_index int64 [2,E]
    const float*     wgt   = (const float*)d_in[2];
    const float*     W     = (const float*)d_in[3];
    const float*     b     = (const float*)d_in[4];
    const float*     alpha = (const float*)d_in[5];
    float*           out   = (float*)d_out;

    // Workspace layout (256B aligned offsets):
    //   xh  : N*DIN f16  = 12,800,000 B  @ 0
    //   wh  : DH*DIN f16 =     32,768 B  @ 12,800,000
    //   deg : N f32      =    200,000 B  @ 12,832,768   (becomes dinv in place)
    //   h   : N*DH f32   = 25,600,000 B  @ 13,032,960
    char* ws = (char*)d_ws;
    _Float16* xh  = (_Float16*)(ws);
    _Float16* wh  = (_Float16*)(ws + 12800000);
    float*    deg = (float*)   (ws + 12832768);
    float*    h   = (float*)   (ws + 13032960);

    const int TPB = 256;

    // 1) f32 -> f16 conversion of x and W
    {
        int total = NNODES * DIN + DH * DIN;
        convert_f16_kernel<<<(total + TPB - 1) / TPB, TPB, 0, stream>>>(x, W, xh, wh);
    }

    // 2) WMMA GEMM: 3125 M-tiles x 8 N-tile waves per block
    gemm_wmma_kernel<<<NNODES / 16, TPB, 0, stream>>>(xh, wh, h);

    // 3) degree -> dinv
    deg_init_kernel<<<(NNODES + TPB - 1) / TPB, TPB, 0, stream>>>(deg);
    deg_scatter_kernel<<<(NEDGES + TPB - 1) / TPB, TPB, 0, stream>>>(ei, wgt, deg);
    dinv_kernel<<<(NNODES + TPB - 1) / TPB, TPB, 0, stream>>>(deg);

    // 4) out = bias + self-loop contribution
    out_init_kernel<<<(NNODES * DH + TPB - 1) / TPB, TPB, 0, stream>>>(h, deg, b, out);

    // 5) edge gather/scale/scatter: one wave per edge
    {
        long long threads = (long long)NEDGES * 32;
        int blocks = (int)((threads + TPB - 1) / TPB);
        edge_scatter_kernel<<<blocks, TPB, 0, stream>>>(ei, wgt, deg, h, out);
    }

    // 6) PReLU
    prelu_kernel<<<(NNODES * DH + TPB - 1) / TPB, TPB, 0, stream>>>(alpha, out);
}
